// FNetLayer_9912784519360
// MI455X (gfx1250) — matrix operand order repaired and checked by reading, not compiled, not run on hardware
//
#include <hip/hip_runtime.h>
#include <hip/hip_bf16.h>
#include <cmath>

typedef __bf16 bf16;
typedef __attribute__((ext_vector_type(16))) __bf16 v16bf;
typedef __attribute__((ext_vector_type(8)))  __bf16 v8bf;
typedef __attribute__((ext_vector_type(8)))  float  v8f;
typedef int v4i __attribute__((vector_size(4 * sizeof(int))));

#define FN_B 4
#define FN_S 4096
#define FN_H 1024
#define FN_I 4096

#define BM 128
#define BN 128
#define BK 32
#define KPAD 40   // BK + 8 halves padding -> 80B row stride, 16B aligned

// --------------------------------------------------------------------------
// CDNA5 async global->LDS path (ASYNCcnt), with safe fallback
// --------------------------------------------------------------------------
#if defined(__has_builtin)
#  if __has_builtin(__builtin_amdgcn_global_load_async_to_lds_b128)
#    define USE_ASYNC_LDS 1
#  endif
#endif
#ifndef USE_ASYNC_LDS
#  define USE_ASYNC_LDS 0
#endif

#if USE_ASYNC_LDS
typedef __attribute__((address_space(1))) v4i* gv4i_ptr;
typedef __attribute__((address_space(3))) v4i* lv4i_ptr;
__device__ __forceinline__ void async_ld_b128(const bf16* g, bf16* l) {
    __builtin_amdgcn_global_load_async_to_lds_b128(
        (gv4i_ptr)g, (lv4i_ptr)l, 0, 0);
}
__device__ __forceinline__ void wait_async0() {
#if __has_builtin(__builtin_amdgcn_s_wait_asynccnt)
    __builtin_amdgcn_s_wait_asynccnt(0);
#else
    asm volatile("s_wait_asynccnt 0x0" ::: "memory");
#endif
}
#endif

// ---------------------------------------------------------------------------
// DFT cosine matrix: C[k][n] = cos(2*pi*k*n/S) / sqrt(S), bf16
// ---------------------------------------------------------------------------
__global__ __launch_bounds__(256) void build_dft_kernel(bf16* __restrict__ C) {
    const int idx = blockIdx.x * 256 + threadIdx.x;      // < S*S = 16.7M
    const int k = idx >> 12;          // / 4096
    const int n = idx & 4095;         // % 4096
    const unsigned m = ((unsigned)k * (unsigned)n) & 4095u;  // k*n mod S
    const float ang = (float)m * (6.283185307179586f / 4096.0f);
    C[idx] = (bf16)(__cosf(ang) * 0.015625f);            // 1/sqrt(4096)=1/64
}

// ---------------------------------------------------------------------------
// Transpose-cast f32 [rows][cols] -> bf16 [cols][rows], batched over blockIdx.y
// ---------------------------------------------------------------------------
__global__ __launch_bounds__(256)
void transpose_cast_kernel(const float* __restrict__ in, bf16* __restrict__ out,
                           int rows, int cols) {
    const int b = blockIdx.y;
    const size_t idx = (size_t)blockIdx.x * 256 + threadIdx.x;
    const size_t nelem = (size_t)rows * cols;
    if (idx >= nelem) return;
    const int r = (int)(idx / cols);
    const int c = (int)(idx % cols);
    const float* src = in  + (size_t)b * nelem;
    bf16*       dst = out + (size_t)b * nelem;
    dst[(size_t)c * rows + r] = (bf16)src[idx];
}

// ---------------------------------------------------------------------------
// bf16 WMMA GEMM: out[M,N] = A[M,K] @ Bt[N,K]^T  (+ epilogue)
//   MODE 0: outF = acc + resid                     (f32 out)
//   MODE 1: outB = gelu_exact(acc + bias)          (bf16 out)
//   MODE 2: outF = acc + bias + resid              (f32 out)
// ---------------------------------------------------------------------------
template <int MODE>
__global__ __launch_bounds__(256)
void gemm_bf16_wmma(const bf16* __restrict__ A,  int lda,
                    const bf16* __restrict__ Bt, int ldb,
                    const float* __restrict__ bias,
                    const float* __restrict__ resid, int ldr,
                    float* __restrict__ outF, bf16* __restrict__ outB, int ldo,
                    int K)
{
    __shared__ bf16 As[2][BM][KPAD];
    __shared__ bf16 Bs[2][BN][KPAD];

    const int tid  = threadIdx.x;
    const int lane = tid & 31;
    const int wave = tid >> 5;
    const int wm   = wave & 1;        // 2 M-groups of 64 rows
    const int wn   = wave >> 1;       // 4 N-groups of 32 cols
    const int hs   = lane >> 4;       // half-wave select
    const int l    = lane & 15;

    const int blockN = blockIdx.x * BN;
    const int blockM = blockIdx.y * BM;

    // global->LDS mapping: each thread moves 16 halves of A and 16 of B
    const int ltr = tid >> 1;           // 0..127 tile row
    const int ltk = (tid & 1) << 4;     // 0 or 16

    v8f acc[4][2];
#pragma unroll
    for (int mi = 0; mi < 4; ++mi)
#pragma unroll
        for (int ni = 0; ni < 2; ++ni)
            acc[mi][ni] = (v8f){0.f,0.f,0.f,0.f,0.f,0.f,0.f,0.f};

    const int nK = K / BK;
    const bf16* aRow = A  + (size_t)(blockM + ltr) * lda + ltk;
    const bf16* bRow = Bt + (size_t)(blockN + ltr) * ldb + ltk;

#if USE_ASYNC_LDS
    // prologue: DMA tile 0 straight into LDS
    async_ld_b128(aRow,     &As[0][ltr][ltk]);
    async_ld_b128(aRow + 8, &As[0][ltr][ltk + 8]);
    async_ld_b128(bRow,     &Bs[0][ltr][ltk]);
    async_ld_b128(bRow + 8, &Bs[0][ltr][ltk + 8]);
    wait_async0();
    __syncthreads();
#else
    uint4 ra0, ra1, rb0, rb1;
    ra0 = *(const uint4*)(aRow);
    ra1 = *(const uint4*)(aRow + 8);
    rb0 = *(const uint4*)(bRow);
    rb1 = *(const uint4*)(bRow + 8);
    *(uint4*)&As[0][ltr][ltk]     = ra0;
    *(uint4*)&As[0][ltr][ltk + 8] = ra1;
    *(uint4*)&Bs[0][ltr][ltk]     = rb0;
    *(uint4*)&Bs[0][ltr][ltk + 8] = rb1;
    __syncthreads();
#endif

    for (int kt = 0; kt < nK; ++kt) {
        const int cur = kt & 1, nxt = cur ^ 1;
        const bool has_next = (kt + 1) < nK;

#if USE_ASYNC_LDS
        // Issue DMA for the next tile up front: the end-of-previous-iteration
        // barrier already guarantees every wave finished reading buffer `nxt`,
        // so the async writes overlap this iteration's WMMA burst.
        if (has_next) {
            const int kb = (kt + 1) * BK;
            async_ld_b128(aRow + kb,     &As[nxt][ltr][ltk]);
            async_ld_b128(aRow + kb + 8, &As[nxt][ltr][ltk + 8]);
            async_ld_b128(bRow + kb,     &Bs[nxt][ltr][ltk]);
            async_ld_b128(bRow + kb + 8, &Bs[nxt][ltr][ltk + 8]);
        }
#else
        if (has_next) {
            const int kb = (kt + 1) * BK;
            __builtin_prefetch(aRow + kb + BK, 0, 1);   // global_prefetch_b8
            __builtin_prefetch(bRow + kb + BK, 0, 1);
            ra0 = *(const uint4*)(aRow + kb);
            ra1 = *(const uint4*)(aRow + kb + 8);
            rb0 = *(const uint4*)(bRow + kb);
            rb1 = *(const uint4*)(bRow + kb + 8);
        }
#endif

        // fragments per ISA 7.12.2 bf16 layouts
        v16bf afrag[4], bfrag[2];
#pragma unroll
        for (int mi = 0; mi < 4; ++mi) {
            const bf16* p = &As[cur][wm * 64 + mi * 16 + l][hs * 8];
            v8bf lo = *(const v8bf*)(p);        // K = hs*8 .. +7
            v8bf hi = *(const v8bf*)(p + 16);   // K = hs*8+16 .. +23
            afrag[mi] = __builtin_shufflevector(lo, hi,
                0,1,2,3,4,5,6,7,8,9,10,11,12,13,14,15);
        }
#pragma unroll
        for (int ni = 0; ni < 2; ++ni) {
            const bf16* p = &Bs[cur][wn * 32 + ni * 16 + l][hs * 8];
            v8bf lo = *(const v8bf*)(p);        // K = hs*8 .. +7
            v8bf hi = *(const v8bf*)(p + 16);   // K = hs*8+16 .. +23
            bfrag[ni] = __builtin_shufflevector(lo, hi,
                0,1,2,3,4,5,6,7,8,9,10,11,12,13,14,15);
        }

#pragma unroll
        for (int mi = 0; mi < 4; ++mi)
#pragma unroll
            for (int ni = 0; ni < 2; ++ni)
                acc[mi][ni] = __builtin_amdgcn_wmma_f32_16x16x32_bf16(
                    false, afrag[mi], false, bfrag[ni],
                    (short)0, acc[mi][ni], false, false);

        if (has_next) {
#if USE_ASYNC_LDS
            wait_async0();        // this wave's 4 DMAs into `nxt` landed
            __syncthreads();      // everyone else's too
#else
            *(uint4*)&As[nxt][ltr][ltk]     = ra0;
            *(uint4*)&As[nxt][ltr][ltk + 8] = ra1;
            *(uint4*)&Bs[nxt][ltr][ltk]     = rb0;
            *(uint4*)&Bs[nxt][ltr][ltk + 8] = rb1;
            __syncthreads();
#endif
        }
    }

    // epilogue (D layout: lanes<16 -> M=r, lanes>=16 -> M=r+8; N=l)
#pragma unroll
    for (int mi = 0; mi < 4; ++mi) {
#pragma unroll
        for (int ni = 0; ni < 2; ++ni) {
#pragma unroll
            for (int r = 0; r < 8; ++r) {
                const int grow = blockM + wm * 64 + mi * 16 + r + hs * 8;
                const int gcol = blockN + wn * 32 + ni * 16 + l;
                float v = acc[mi][ni][r];
                if (MODE == 0) {
                    v += resid[(size_t)grow * ldr + gcol];
                    outF[(size_t)grow * ldo + gcol] = v;
                } else if (MODE == 1) {
                    v += bias[gcol];
                    v = 0.5f * v * (1.0f + erff(v * 0.7071067811865476f));
                    outB[(size_t)grow * ldo + gcol] = (bf16)v;
                } else {
                    v += bias[gcol] + resid[(size_t)grow * ldr + gcol];
                    outF[(size_t)grow * ldo + gcol] = v;
                }
            }
        }
    }
}

// ---------------------------------------------------------------------------
// LayerNorm over H=1024 per row, 256 threads/row; optional bf16 copy out.
// ---------------------------------------------------------------------------
__global__ __launch_bounds__(256)
void layernorm_kernel(const float* __restrict__ in,
                      const float* __restrict__ gamma,
                      const float* __restrict__ beta,
                      float* __restrict__ outF, bf16* __restrict__ outB)
{
    __shared__ float ssum[256];
    __shared__ float ssq[256];
    const int tid = threadIdx.x;
    const float* x = in + (size_t)blockIdx.x * FN_H;

    float v[4], s = 0.f, s2 = 0.f;
#pragma unroll
    for (int i = 0; i < 4; ++i) {
        v[i] = x[tid + i * 256];
        s  += v[i];
        s2 += v[i] * v[i];
    }
    ssum[tid] = s; ssq[tid] = s2;
    __syncthreads();
    for (int off = 128; off > 0; off >>= 1) {
        if (tid < off) { ssum[tid] += ssum[tid + off]; ssq[tid] += ssq[tid + off]; }
        __syncthreads();
    }
    const float mean = ssum[0] * (1.0f / FN_H);
    const float var  = ssq[0] * (1.0f / FN_H) - mean * mean;
    const float rstd = rsqrtf(var + 1e-12f);

    float* yF = outF + (size_t)blockIdx.x * FN_H;
    bf16*  yB = outB ? outB + (size_t)blockIdx.x * FN_H : nullptr;
#pragma unroll
    for (int i = 0; i < 4; ++i) {
        const int c = tid + i * 256;
        const float y = (v[i] - mean) * rstd * gamma[c] + beta[c];
        yF[c] = y;
        if (yB) yB[c] = (bf16)y;
    }
}

// ---------------------------------------------------------------------------
extern "C" void kernel_launch(void* const* d_in, const int* in_sizes, int n_in,
                              void* d_out, int out_size, void* d_ws, size_t ws_size,
                              hipStream_t stream) {
    (void)in_sizes; (void)n_in; (void)out_size; (void)ws_size;
    const float* hidden = (const float*)d_in[0];
    const float* ln1_g  = (const float*)d_in[1];
    const float* ln1_b  = (const float*)d_in[2];
    const float* W1     = (const float*)d_in[3];
    const float* b1     = (const float*)d_in[4];
    const float* W2     = (const float*)d_in[5];
    const float* b2     = (const float*)d_in[6];
    const float* ln2_g  = (const float*)d_in[7];
    const float* ln2_b  = (const float*)d_in[8];
    float* out = (float*)d_out;

    char* ws = (char*)d_ws;
    const size_t SS   = (size_t)FN_S * FN_S;          // 16.7M
    const size_t BSH  = (size_t)FN_B * FN_S * FN_H;   // 16.7M
    const size_t BSI  = (size_t)FN_B * FN_S * FN_I;   // 67M
    bf16*  Cdft = (bf16*) (ws);                          size_t o = SS * 2;
    bf16*  xT   = (bf16*) (ws + o);                      o += BSH * 2;
    float* y    = (float*)(ws + o);                      o += BSH * 4;
    float* x1   = (float*)(ws + o);                      o += BSH * 4;
    bf16*  x1b  = (bf16*) (ws + o);                      o += BSH * 2;
    bf16*  W1t  = (bf16*) (ws + o);                      o += (size_t)FN_H * FN_I * 2;
    bf16*  W2t  = (bf16*) (ws + o);                      o += (size_t)FN_I * FN_H * 2;
    bf16*  hb   = (bf16*) (ws + o);                      o += BSI * 2;

    const int M = FN_B * FN_S;                         // 16384 rows

    // 1) DFT cosine matrix
    build_dft_kernel<<<(int)(SS / 256), 256, 0, stream>>>(Cdft);

    // 2) casts/transposes to bf16
    {
        dim3 g((unsigned)(((size_t)FN_S * FN_H + 255) / 256), FN_B);
        transpose_cast_kernel<<<g, 256, 0, stream>>>(hidden, xT, FN_S, FN_H);
    }
    {
        dim3 g((unsigned)(((size_t)FN_H * FN_I + 255) / 256), 1);
        transpose_cast_kernel<<<g, 256, 0, stream>>>(W1, W1t, FN_H, FN_I);
        transpose_cast_kernel<<<g, 256, 0, stream>>>(W2, W2t, FN_I, FN_H);
    }

    // 3) Fourier mixing as GEMM + residual:  y[b] = Cdft @ x[b] + x[b]
    for (int b = 0; b < FN_B; ++b) {
        dim3 grid(FN_H / BN, FN_S / BM);
        gemm_bf16_wmma<0><<<grid, 256, 0, stream>>>(
            Cdft, FN_S,
            xT + (size_t)b * FN_H * FN_S, FN_S,
            nullptr,
            hidden + (size_t)b * FN_S * FN_H, FN_H,
            y + (size_t)b * FN_S * FN_H, nullptr, FN_H,
            FN_S);
    }

    // 4) LayerNorm1 -> x1 (f32) + x1b (bf16)
    layernorm_kernel<<<M, 256, 0, stream>>>(y, ln1_g, ln1_b, x1, x1b);

    // 5) h = gelu(x1 @ W1 + b1)  -> bf16
    {
        dim3 grid(FN_I / BN, M / BM);
        gemm_bf16_wmma<1><<<grid, 256, 0, stream>>>(
            x1b, FN_H, W1t, FN_H, b1, nullptr, 0,
            nullptr, hb, FN_I, FN_H);
    }

    // 6) y = h @ W2 + b2 + x1
    {
        dim3 grid(FN_H / BN, M / BM);
        gemm_bf16_wmma<2><<<grid, 256, 0, stream>>>(
            hb, FN_I, W2t, FN_I, b2, x1, FN_H,
            y, nullptr, FN_H, FN_I);
    }

    // 7) LayerNorm2 -> out (f32)
    layernorm_kernel<<<M, 256, 0, stream>>>(y, ln2_g, ln2_b, out, nullptr);
}
